// TR_Linear_21182778704893
// MI455X (gfx1250) — matrix operand order repaired and checked by reading, not compiled, not run on hardware
//
#include <hip/hip_runtime.h>
#include <hip/hip_bf16.h>

// ---------------------------------------------------------------------------
// TR-Linear on gfx1250 (MI455X):
//   out(8192x4096) = x(8192x1024) @ W^T + bias,  W assembled from 4 TR cores.
// Main GEMM: v_wmma_f32_16x16x32_f16 with hi/lo f16 split (AhBh+AhBl+AlBh,
// f32 accumulate) => ~fp32 accuracy at f16 WMMA rate.  Tiles staged with
// GLOBAL_LOAD_ASYNC_TO_LDS_B128 + double-buffered LDS when available.
// ---------------------------------------------------------------------------

typedef __attribute__((ext_vector_type(16))) _Float16 v16h;
typedef __attribute__((ext_vector_type(8)))  _Float16 v8h;
typedef __attribute__((ext_vector_type(8)))  float    v8f;
typedef __attribute__((ext_vector_type(4)))  int      v4i;

// pointer-to-(v4i in AS1/AS3); builtin takes (global v4i*, lds v4i*, imm, imm)
typedef __attribute__((address_space(1))) v4i* gptr_v4i;
typedef __attribute__((address_space(3))) v4i* lptr_v4i;

#define M_DIM 8192
#define K_DIM 1024
#define N_DIM 4096
#define R_DIM 16     // TR rank
#define AB_DIM 256   // 16*16 fused rank pair

#if defined(__has_builtin)
#if __has_builtin(__builtin_amdgcn_global_load_async_to_lds_b128) && \
    __has_builtin(__builtin_amdgcn_s_wait_asynccnt)
#define ASYNC_LDS 1
#endif
#endif

// 16-byte global -> LDS copy (async DMA when the toolchain exposes it).
__device__ __forceinline__ void lds_cp16(_Float16* dst, const _Float16* src) {
#ifdef ASYNC_LDS
    __builtin_amdgcn_global_load_async_to_lds_b128(
        (gptr_v4i)src, (lptr_v4i)dst, 0, 0);
#else
    *(uint4*)dst = *(const uint4*)src;
#endif
}

__device__ __forceinline__ void wait_async_lds() {
#ifdef ASYNC_LDS
    __builtin_amdgcn_s_wait_asynccnt(0);
#endif
}

// ---------------------------------------------------------------------------
// Kernel 1: M1[ii, a*16+b] = sum_k core0[a,i0,k] * core1[k,i1,b]
//   core0: (16,32,16), core1: (16,32,16); ii = i0*32+i1.  262144 outputs.
// ---------------------------------------------------------------------------
__global__ __launch_bounds__(256) void tr_build_m1(const float* __restrict__ core0,
                                                   const float* __restrict__ core1,
                                                   float* __restrict__ m1) {
    const int idx = blockIdx.x * 256 + threadIdx.x;     // [0, 1024*256)
    const int ii = idx >> 8;
    const int ab = idx & 255;
    const int a = ab >> 4, b = ab & 15;
    const int i0 = ii >> 5, i1 = ii & 31;
    const float* c0 = core0 + (a * 32 + i0) * 16;       // contiguous in k
    float acc = 0.f;
#pragma unroll
    for (int k = 0; k < R_DIM; ++k)
        acc += c0[k] * core1[(k * 32 + i1) * 16 + b];
    m1[idx] = acc;
}

// ---------------------------------------------------------------------------
// Kernel 2: M2T[oo, a*16+b] = sum_c core2[b,o0,c] * core3[c,o1,a]
//   core2: (16,64,16), core3: (16,64,16); oo = o0*64+o1.  1048576 outputs.
// ---------------------------------------------------------------------------
__global__ __launch_bounds__(256) void tr_build_m2t(const float* __restrict__ core2,
                                                    const float* __restrict__ core3,
                                                    float* __restrict__ m2t) {
    const int idx = blockIdx.x * 256 + threadIdx.x;     // [0, 4096*256)
    const int oo = idx >> 8;
    const int ab = idx & 255;
    const int a = ab >> 4, b = ab & 15;
    const int o0 = oo >> 6, o1 = oo & 63;
    const float* c2 = core2 + (b * 64 + o0) * 16;       // contiguous in c
    float acc = 0.f;
#pragma unroll
    for (int c = 0; c < R_DIM; ++c)
        acc += c2[c] * core3[(c * 64 + o1) * 16 + a];
    m2t[idx] = acc;
}

// ---------------------------------------------------------------------------
// Kernel 3: W[ii,oo] = M1(1024x256) @ M2T^T(256x4096), exact fp32, written as
// hi/lo f16 pair at flat index ii*4096+oo (== r*1024+c of the (4096,1024)
// view the main GEMM consumes). 64x64 output tile per block, K chunked by 64.
// ---------------------------------------------------------------------------
__global__ __launch_bounds__(256) void tr_build_w(const float* __restrict__ m1,
                                                  const float* __restrict__ m2t,
                                                  _Float16* __restrict__ wh,
                                                  _Float16* __restrict__ wl) {
    __shared__ float sM1[64][72];   // [ii][k], padded rows (288B, 16B aligned)
    __shared__ float sM2[64][72];   // [oo][k]
    const int t  = threadIdx.x;
    const int tx = t & 15;          // oo group (4 each)
    const int ty = t >> 4;          // ii group (4 each)
    const int oob = blockIdx.x * 64;
    const int iib = blockIdx.y * 64;

    float acc[4][4];
#pragma unroll
    for (int i = 0; i < 4; ++i)
#pragma unroll
        for (int j = 0; j < 4; ++j) acc[i][j] = 0.f;

    for (int kb = 0; kb < AB_DIM; kb += 64) {
#pragma unroll
        for (int p = 0; p < 4; ++p) {               // 1024 float4-chunks / 256 thr
            const int c = t + p * 256;
            const int row = c >> 4, seg = c & 15;
            *(float4*)&sM1[row][seg * 4] = *(const float4*)&m1 [(iib + row) * AB_DIM + kb + seg * 4];
            *(float4*)&sM2[row][seg * 4] = *(const float4*)&m2t[(oob + row) * AB_DIM + kb + seg * 4];
        }
        __syncthreads();
#pragma unroll 4
        for (int k = 0; k < 64; ++k) {
            float av[4], bv[4];
#pragma unroll
            for (int i = 0; i < 4; ++i) av[i] = sM1[ty * 4 + i][k];
#pragma unroll
            for (int j = 0; j < 4; ++j) bv[j] = sM2[tx * 4 + j][k];
#pragma unroll
            for (int i = 0; i < 4; ++i)
#pragma unroll
                for (int j = 0; j < 4; ++j) acc[i][j] = fmaf(av[i], bv[j], acc[i][j]);
        }
        __syncthreads();
    }
#pragma unroll
    for (int i = 0; i < 4; ++i)
#pragma unroll
        for (int j = 0; j < 4; ++j) {
            const int ii = iib + ty * 4 + i;
            const int oo = oob + tx * 4 + j;
            const float w = acc[i][j];
            const _Float16 h = (_Float16)w;
            wh[ii * N_DIM + oo] = h;
            wl[ii * N_DIM + oo] = (_Float16)(w - (float)h);
        }
}

// ---------------------------------------------------------------------------
// Kernel 4: split x into hi/lo f16 halves.
// ---------------------------------------------------------------------------
__global__ __launch_bounds__(256) void tr_split_x(const float* __restrict__ x,
                                                  _Float16* __restrict__ xh,
                                                  _Float16* __restrict__ xl,
                                                  int n) {
    int i = blockIdx.x * 256 + threadIdx.x;
    const int stride = gridDim.x * 256;
    for (; i < n; i += stride) {
        const float v = x[i];
        const _Float16 h = (_Float16)v;
        xh[i] = h;
        xl[i] = (_Float16)(v - (float)h);
    }
}

// ---------------------------------------------------------------------------
// Kernel 5: main GEMM.  Block = 128x128 tile, 256 threads = 8 waves, wave w
// owns a 32x64 sub-tile: (wm = w&3) -> m offset, (wn = w>>2) -> n offset,
// 2x4 C fragments (16x16, v8f).  BK = 32, LDS rows padded to 40 f16.
// Double-buffered: async-load tile k+1 while issuing 24 WMMAs for tile k.
// ---------------------------------------------------------------------------
#define LDA 40        // padded f16 row stride (80B: 16B-aligned, spreads banks)
#define NKT (K_DIM / 32)

__global__ __launch_bounds__(256) void tr_wmma_gemm(const _Float16* __restrict__ xh,
                                                    const _Float16* __restrict__ xl,
                                                    const _Float16* __restrict__ wh,
                                                    const _Float16* __restrict__ wl,
                                                    const float* __restrict__ bias,
                                                    float* __restrict__ out) {
    __shared__ _Float16 sAh[2][128 * LDA];
    __shared__ _Float16 sAl[2][128 * LDA];
    __shared__ _Float16 sBh[2][128 * LDA];
    __shared__ _Float16 sBl[2][128 * LDA];

    const int t     = threadIdx.x;
    const int lane  = t & 31;
    const int wave  = t >> 5;
    const int wm    = wave & 3;          // 0..3 -> m offset wm*32
    const int wn    = wave >> 2;         // 0..1 -> n offset wn*64
    const int l15   = lane & 15;
    const int lhalf = lane >> 4;         // 0: lanes 0-15, 1: lanes 16-31

    const int mb = blockIdx.y * 128;
    const int nb = blockIdx.x * 128;

    v8f acc[2][4] = {};

    union Frag { v16h v; v8h h[2]; };

    // ---- issue global->LDS copies for one 128x32 k-tile (8 x 16B/thread) ----
    auto issue_tile = [&](int buf, int kt) {
        const int koff = kt * 32;
#pragma unroll
        for (int p = 0; p < 2; ++p) {
            const int c   = t + p * 256;
            const int row = c >> 2;
            const int seg = c & 3;
            const int ga  = (mb + row) * K_DIM + koff + seg * 8;
            const int gb  = (nb + row) * K_DIM + koff + seg * 8;
            const int la  = row * LDA + seg * 8;
            lds_cp16(&sAh[buf][la], &xh[ga]);
            lds_cp16(&sAl[buf][la], &xl[ga]);
            lds_cp16(&sBh[buf][la], &wh[gb]);
            lds_cp16(&sBl[buf][la], &wl[gb]);
        }
    };

    issue_tile(0, 0);

    for (int kt = 0; kt < NKT; ++kt) {
        const int cur = kt & 1;
        wait_async_lds();        // this wave's tile-kt DMA complete
        __syncthreads();         // all waves' tile-kt data resident; buf[!cur] free
        if (kt + 1 < NKT) issue_tile(cur ^ 1, kt + 1);   // overlap with WMMAs

        // ---- LDS -> fragments (ds_load_b128 pairs, wave32 WMMA layouts) ----
        Frag ah[2], al[2], bh[4], bl[4];
#pragma unroll
        for (int m = 0; m < 2; ++m) {
            const int row = (wm * 32 + m * 16 + l15) * LDA;
            // A 16-bit 16x32: lanes0-15 v0-3=K0..7, v4-7=K16..23; lanes16-31 +8
            ah[m].h[0] = *(const v8h*)&sAh[cur][row + lhalf * 8];
            ah[m].h[1] = *(const v8h*)&sAh[cur][row + 16 + lhalf * 8];
            al[m].h[0] = *(const v8h*)&sAl[cur][row + lhalf * 8];
            al[m].h[1] = *(const v8h*)&sAl[cur][row + 16 + lhalf * 8];
        }
#pragma unroll
        for (int n = 0; n < 4; ++n) {
            const int row = (wn * 64 + n * 16 + l15) * LDA;
            // B 16-bit 32x16: lanes0-15 K0..7, K16..23; lanes16-31 +8
            bh[n].h[0] = *(const v8h*)&sBh[cur][row + lhalf * 8];
            bh[n].h[1] = *(const v8h*)&sBh[cur][row + 16 + lhalf * 8];
            bl[n].h[0] = *(const v8h*)&sBl[cur][row + lhalf * 8];
            bl[n].h[1] = *(const v8h*)&sBl[cur][row + 16 + lhalf * 8];
        }

        // ---- 2x4 fragments x 3 split-products = 24 WMMAs per k-step ----
#pragma unroll
        for (int m = 0; m < 2; ++m)
#pragma unroll
            for (int n = 0; n < 4; ++n) {
                acc[m][n] = __builtin_amdgcn_wmma_f32_16x16x32_f16(
                    false, ah[m].v, false, bh[n].v, (short)0, acc[m][n], false, false);
                acc[m][n] = __builtin_amdgcn_wmma_f32_16x16x32_f16(
                    false, ah[m].v, false, bl[n].v, (short)0, acc[m][n], false, false);
                acc[m][n] = __builtin_amdgcn_wmma_f32_16x16x32_f16(
                    false, al[m].v, false, bh[n].v, (short)0, acc[m][n], false, false);
            }
    }

    // ---- epilogue: C layout = lanes0-15 VGPRj->M=j, lanes16-31 -> M=8+j ----
#pragma unroll
    for (int m = 0; m < 2; ++m)
#pragma unroll
        for (int n = 0; n < 4; ++n) {
            const int col = nb + wn * 64 + n * 16 + l15;
            const float bv = bias[col];
#pragma unroll
            for (int j = 0; j < 8; ++j) {
                const int row = mb + wm * 32 + m * 16 + lhalf * 8 + j;
                out[row * N_DIM + col] = acc[m][n][j] + bv;
            }
        }
}

// ---------------------------------------------------------------------------
// Launch.  Workspace layout (53 MB):
//   xh 16MB | xl 16MB | wh 8MB | wl 8MB | m1 1MB | m2t 4MB
// ---------------------------------------------------------------------------
extern "C" void kernel_launch(void* const* d_in, const int* in_sizes, int n_in,
                              void* d_out, int out_size, void* d_ws, size_t ws_size,
                              hipStream_t stream) {
    const float* x     = (const float*)d_in[0];
    const float* core0 = (const float*)d_in[1];
    const float* core1 = (const float*)d_in[2];
    const float* core2 = (const float*)d_in[3];
    const float* core3 = (const float*)d_in[4];
    const float* bias  = (const float*)d_in[5];
    float* out = (float*)d_out;

    _Float16* xh = (_Float16*)d_ws;
    _Float16* xl = xh + (size_t)M_DIM * K_DIM;
    _Float16* wh = xl + (size_t)M_DIM * K_DIM;
    _Float16* wl = wh + (size_t)N_DIM * K_DIM;
    float*    m1  = (float*)(wl + (size_t)N_DIM * K_DIM);
    float*    m2t = m1 + (size_t)K_DIM * AB_DIM;

    tr_build_m1 <<<(K_DIM * AB_DIM) / 256, 256, 0, stream>>>(core0, core1, m1);
    tr_build_m2t<<<(N_DIM * AB_DIM) / 256, 256, 0, stream>>>(core2, core3, m2t);
    tr_build_w  <<<dim3(N_DIM / 64, K_DIM / 64), 256, 0, stream>>>(m1, m2t, wh, wl);
    tr_split_x  <<<4096, 256, 0, stream>>>(x, xh, xl, M_DIM * K_DIM);
    tr_wmma_gemm<<<dim3(N_DIM / 128, M_DIM / 128), 256, 0, stream>>>(xh, xl, wh, wl, bias, out);
}
